// Decoder_29042568856181
// MI455X (gfx1250) — compile-verified, matrix operand order checked by reading, not run on hardware
//
#include <hip/hip_runtime.h>
#include <hip/hip_bf16.h>

typedef __attribute__((ext_vector_type(16))) __bf16          v16bf;
typedef __attribute__((ext_vector_type(16))) unsigned short  v16us;
typedef __attribute__((ext_vector_type(8)))  unsigned short  v8us;
typedef __attribute__((ext_vector_type(8)))  float           v8f;

#define T_STEPS 256
#define BATCH   1024
#define BT      16

// ---- dynamic LDS layout (bytes) ----
#define XS_OFF    0                     // float  [256][16]            16384
#define G3_OFF    16384                 // float  [16][17]              1088
#define H1_OFF    17472                 // ushort [16][72]              2304
#define H1M_OFF   19776                 // ushort [16][72]              2304
#define H2_OFF    22080                 // ushort [16][136]             4352
#define H2M_OFF   26432                 // ushort [16][136]             4352
#define W1_OFF    30784                 // ushort [256][72]  (Whh1)    36864
#define W2I_OFF   67648                 // ushort [512][72]  (Wih2)    73728
#define W2H_OFF   141376                // ushort [512][136] (Whh2)   139264
#define W3_OFF    280640                // ushort [16][136]  (Wih3 pad) 4352
#define SMEM_BYTES 284992               // 278.3 KB < 320 KB/WG

__device__ __forceinline__ unsigned short f2bf(float f) {
  unsigned u = __builtin_bit_cast(unsigned, f);
  u += 0x7FFFu + ((u >> 16) & 1u);     // round-to-nearest-even
  return (unsigned short)(u >> 16);
}

// gfx1250 has a hardware v_tanh_f32 (TRANS32). Use it for both activations.
#if __has_builtin(__builtin_amdgcn_tanhf)
__device__ __forceinline__ float tanh_hw(float x) { return __builtin_amdgcn_tanhf(x); }
#else
__device__ __forceinline__ float tanh_hw(float x) {
  float e = __expf(-2.0f * x);                       // v_exp_f32
  return (1.0f - e) * __builtin_amdgcn_rcpf(1.0f + e); // v_rcp_f32, no IEEE div expansion
}
#endif
__device__ __forceinline__ float sigf(float x) { return 0.5f + 0.5f * tanh_hw(0.5f * x); }

// Load one 16x32 bf16 WMMA fragment from LDS ([M][K] for A, [N][K] for B).
// lanes 0-15: row=lane, K = {0..7, 16..23}; lanes 16-31: row=lane-16, K = {8..15, 24..31}
__device__ __forceinline__ v16bf frag_ld(const unsigned short* p, int stride, int lane) {
  int row = lane & 15;
  int kb  = (lane >> 4) << 3;          // 0 or 8
  const unsigned short* q = p + row * stride + kb;
  v8us lo = *(const v8us*)q;           // K = kb .. kb+7
  v8us hi = *(const v8us*)(q + 16);    // K = kb+16 .. kb+23
  v16us t = __builtin_shufflevector(lo, hi, 0,1,2,3,4,5,6,7,8,9,10,11,12,13,14,15);
  return __builtin_bit_cast(v16bf, t);
}

__global__ __launch_bounds__(256)
void lstm3_fused(const float* __restrict__ x,
                 const float* __restrict__ Wih1, const float* __restrict__ Whh1,
                 const float* __restrict__ b1,
                 const float* __restrict__ Wih2, const float* __restrict__ Whh2,
                 const float* __restrict__ b2,
                 const float* __restrict__ Wih3, const float* __restrict__ Whh3,
                 const float* __restrict__ b3,
                 const float* __restrict__ mask1, const float* __restrict__ mask2,
                 const float* __restrict__ mask3,
                 float* __restrict__ out) {
  extern __shared__ __align__(16) char smem[];
  float*          xs    = (float*)(smem + XS_OFF);
  float*          g3    = (float*)(smem + G3_OFF);
  unsigned short* h1buf = (unsigned short*)(smem + H1_OFF);
  unsigned short* h1msk = (unsigned short*)(smem + H1M_OFF);
  unsigned short* h2buf = (unsigned short*)(smem + H2_OFF);
  unsigned short* h2msk = (unsigned short*)(smem + H2M_OFF);
  unsigned short* w1    = (unsigned short*)(smem + W1_OFF);
  unsigned short* w2i   = (unsigned short*)(smem + W2I_OFF);
  unsigned short* w2h   = (unsigned short*)(smem + W2H_OFF);
  unsigned short* w3    = (unsigned short*)(smem + W3_OFF);

  const int tid  = threadIdx.x;
  const int wave = tid >> 5;
  const int lane = tid & 31;
  const int b0   = blockIdx.x * BT;
  const int nl   = lane & 15;
  const int hi8  = (lane >> 4) << 3;   // C/D fragment row offset 0 / 8

  // ---- one-time: stage x slice + bf16 weights into LDS (padded rows) ----
  for (int i = tid; i < T_STEPS * BT; i += 256) {
    int t = i >> 4, m = i & 15;
    xs[i] = x[t * BATCH + b0 + m];
  }
  for (int i = tid; i < 16 * 72;  i += 256) { h1buf[i] = 0; h1msk[i] = 0; }
  for (int i = tid; i < 16 * 136; i += 256) { h2buf[i] = 0; h2msk[i] = 0; }
  for (int i = tid; i < 256 * 64; i += 256)      // Whh1 [256][64] -> stride 72
    w1[(i >> 6) * 72 + (i & 63)] = f2bf(Whh1[i]);
  for (int i = tid; i < 512 * 64; i += 256)      // Wih2 [512][64] -> stride 72
    w2i[(i >> 6) * 72 + (i & 63)] = f2bf(Wih2[i]);
  for (int i = tid; i < 512 * 128; i += 256)     // Whh2 [512][128] -> stride 136
    w2h[(i >> 7) * 136 + (i & 127)] = f2bf(Whh2[i]);
  for (int i = tid; i < 16 * 128; i += 256) {    // Wih3 [4][128] pad N->16, stride 136
    int row = i >> 7, col = i & 127;
    w3[row * 136 + col] = (row < 4) ? f2bf(Wih3[row * 128 + col]) : (unsigned short)0;
  }

  // ---- small per-wave constants (registers) ----
  float b1v[4], w1v[4], b2v[4];
  v8f   m1f, m2f, c1, c2;
  #pragma unroll
  for (int r = 0; r < 8; r++) { c1[r] = 0.f; c2[r] = 0.f; }
  if (wave < 4) {
    #pragma unroll
    for (int g = 0; g < 4; g++) {
      int nt = g * 4 + wave;
      b1v[g] = b1[nt * 16 + nl];
      w1v[g] = Wih1[nt * 16 + nl];
    }
    #pragma unroll
    for (int r = 0; r < 8; r++) m1f[r] = mask1[(b0 + hi8 + r) * 64 + wave * 16 + nl];
  }
  #pragma unroll
  for (int g = 0; g < 4; g++) b2v[g] = b2[(g * 8 + wave) * 16 + nl];
  #pragma unroll
  for (int r = 0; r < 8; r++) m2f[r] = mask2[(b0 + hi8 + r) * 128 + wave * 16 + nl];
  float c3 = 0.f, h3 = 0.f, whh3v[4], b3v[4], m3v = 0.f;
  if (tid < 16) {
    #pragma unroll
    for (int g = 0; g < 4; g++) { whh3v[g] = Whh3[g]; b3v[g] = b3[g]; }
    m3v = mask3[b0 + tid];
  }
  __syncthreads();

  for (int t = 0; t < T_STEPS; t++) {
    // ================= layer 1 (waves 0-3) =================
    v16bf a10, a11;
    if (wave < 4) {
      a10 = frag_ld(h1buf,      72, lane);
      a11 = frag_ld(h1buf + 32, 72, lane);
    }
    __syncthreads();                          // h1buf reads done before rewrite
    if (wave < 4) {
      v8f acc[4];
      #pragma unroll
      for (int g = 0; g < 4; g++) {
        #pragma unroll
        for (int r = 0; r < 8; r++) acc[g][r] = xs[t * 16 + hi8 + r] * w1v[g] + b1v[g];
      }
      #pragma unroll
      for (int kf = 0; kf < 2; kf++) {
        v16bf a = (kf == 0) ? a10 : a11;
        #pragma unroll
        for (int g = 0; g < 4; g++) {
          v16bf wf = frag_ld(w1 + (g * 4 + wave) * 16 * 72 + kf * 32, 72, lane);
          acc[g] = __builtin_amdgcn_wmma_f32_16x16x32_bf16(false, a, false, wf, (short)0, acc[g], false, false);
        }
      }
      #pragma unroll
      for (int r = 0; r < 8; r++) {
        float iv = sigf(acc[0][r]);
        float fv = sigf(acc[1][r]);
        float gv = tanh_hw(acc[2][r]);
        float ov = sigf(acc[3][r]);
        float cv = fv * c1[r] + iv * gv;
        c1[r] = cv;
        float hv = ov * tanh_hw(cv);
        int m = hi8 + r, col = wave * 16 + nl;
        h1buf[m * 72 + col] = f2bf(hv);
        h1msk[m * 72 + col] = f2bf(hv * m1f[r]);
      }
    }
    __syncthreads();
    // ================= layer 2 (all 8 waves) =================
    v16bf a1m0 = frag_ld(h1msk,      72, lane);
    v16bf a1m1 = frag_ld(h1msk + 32, 72, lane);
    v16bf a2[4];
    #pragma unroll
    for (int kf = 0; kf < 4; kf++) a2[kf] = frag_ld(h2buf + kf * 32, 136, lane);
    __syncthreads();                          // h2buf reads done before rewrite
    {
      v8f acc[4];
      #pragma unroll
      for (int g = 0; g < 4; g++) {
        #pragma unroll
        for (int r = 0; r < 8; r++) acc[g][r] = b2v[g];
      }
      #pragma unroll
      for (int kf = 0; kf < 2; kf++) {
        v16bf a = (kf == 0) ? a1m0 : a1m1;
        #pragma unroll
        for (int g = 0; g < 4; g++) {
          v16bf wf = frag_ld(w2i + (g * 8 + wave) * 16 * 72 + kf * 32, 72, lane);
          acc[g] = __builtin_amdgcn_wmma_f32_16x16x32_bf16(false, a, false, wf, (short)0, acc[g], false, false);
        }
      }
      #pragma unroll
      for (int kf = 0; kf < 4; kf++) {
        #pragma unroll
        for (int g = 0; g < 4; g++) {
          v16bf wf = frag_ld(w2h + (g * 8 + wave) * 16 * 136 + kf * 32, 136, lane);
          acc[g] = __builtin_amdgcn_wmma_f32_16x16x32_bf16(false, a2[kf], false, wf, (short)0, acc[g], false, false);
        }
      }
      #pragma unroll
      for (int r = 0; r < 8; r++) {
        float iv = sigf(acc[0][r]);
        float fv = sigf(acc[1][r]);
        float gv = tanh_hw(acc[2][r]);
        float ov = sigf(acc[3][r]);
        float cv = fv * c2[r] + iv * gv;
        c2[r] = cv;
        float hv = ov * tanh_hw(cv);
        int m = hi8 + r, col = wave * 16 + nl;
        h2buf[m * 136 + col] = f2bf(hv);
        h2msk[m * 136 + col] = f2bf(hv * m2f[r]);
      }
    }
    __syncthreads();
    // ================= layer 3 (wave 0 WMMA, threads 0-15 scalar cell) =================
    if (wave == 0) {
      v8f acc;
      #pragma unroll
      for (int r = 0; r < 8; r++) acc[r] = 0.f;
      #pragma unroll
      for (int kf = 0; kf < 4; kf++) {
        v16bf a  = frag_ld(h2msk + kf * 32, 136, lane);
        v16bf wf = frag_ld(w3 + kf * 32, 136, lane);
        acc = __builtin_amdgcn_wmma_f32_16x16x32_bf16(false, a, false, wf, (short)0, acc, false, false);
      }
      #pragma unroll
      for (int r = 0; r < 8; r++) g3[(hi8 + r) * 17 + nl] = acc[r];
    }
    __syncthreads();
    if (tid < 16) {
      float iv = sigf(g3[tid * 17 + 0] + h3 * whh3v[0] + b3v[0]);
      float fv = sigf(g3[tid * 17 + 1] + h3 * whh3v[1] + b3v[1]);
      float gv = tanh_hw(g3[tid * 17 + 2] + h3 * whh3v[2] + b3v[2]);
      float ov = sigf(g3[tid * 17 + 3] + h3 * whh3v[3] + b3v[3]);
      c3 = fv * c3 + iv * gv;
      h3 = ov * tanh_hw(c3);
      out[t * BATCH + b0 + tid] = h3 * m3v;
    }
    __syncthreads();
  }
}

extern "C" void kernel_launch(void* const* d_in, const int* in_sizes, int n_in,
                              void* d_out, int out_size, void* d_ws, size_t ws_size,
                              hipStream_t stream) {
  const float* x     = (const float*)d_in[0];
  const float* Wih1  = (const float*)d_in[1];
  const float* Whh1  = (const float*)d_in[2];
  const float* b1    = (const float*)d_in[3];
  const float* Wih2  = (const float*)d_in[4];
  const float* Whh2  = (const float*)d_in[5];
  const float* b2    = (const float*)d_in[6];
  const float* Wih3  = (const float*)d_in[7];
  const float* Whh3  = (const float*)d_in[8];
  const float* b3    = (const float*)d_in[9];
  const float* mask1 = (const float*)d_in[10];
  const float* mask2 = (const float*)d_in[11];
  const float* mask3 = (const float*)d_in[12];
  float* out = (float*)d_out;

  lstm3_fused<<<BATCH / BT, 256, SMEM_BYTES, stream>>>(
      x, Wih1, Whh1, b1, Wih2, Whh2, b2, Wih3, Whh3, b3,
      mask1, mask2, mask3, out);
}
